// VectorQuantizer_79826262163871
// MI455X (gfx1250) — matrix operand order repaired and checked by reading, not compiled, not run on hardware
//
#include <hip/hip_runtime.h>
#include <math.h>

// ---------------------------------------------------------------------------
// VQ-VAE vector quantizer for MI455X (gfx1250, wave32, WMMA + async-to-LDS).
//   z:   [32, 256, 32, 32] f32      emb: [1024, 256] f32
// scores = ||e||^2 - 2 * (emb x z[b])   per batch (z[b] is [256,1024] row-major)
// GEMM via v_wmma_f32_16x16x32_bf16 with hi/lo bf16 split (3 terms ~ fp32).
// emb tiles double-buffered in LDS via GLOBAL_LOAD_ASYNC_TO_LDS_B128 so the
// L2->LDS copy of the next 32 codes overlaps the WMMA stream (ASYNCcnt).
// ---------------------------------------------------------------------------

#define N_E     1024
#define E_DIM   256
#define BATCH   32
#define HW      1024
#define NCOL    64          // pixel columns per workgroup
#define NITER   32          // 1024 codes / 32 codes per iteration
#define NPIX    32768       // total pixels (BATCH*HW)

typedef __attribute__((ext_vector_type(16))) __bf16 v16bf;
typedef __attribute__((ext_vector_type(8)))  float  v8f;

__device__ __forceinline__ unsigned short f2bf(float f) {
  unsigned u = __float_as_uint(f);
  u += 0x7FFFu + ((u >> 16) & 1u);        // round-to-nearest-even
  return (unsigned short)(u >> 16);
}
__device__ __forceinline__ float bf2f(unsigned short h) {
  return __uint_as_float(((unsigned)h) << 16);
}

// Async global->LDS copy, 16B per lane. VDST = LDS byte address (low 32 bits of
// the generic pointer per the ISA aperture rules), VADDR = 64-bit global addr.
__device__ __forceinline__ void async_ld_b128(void* lds_dst, const void* gsrc) {
  unsigned lds_off = (unsigned)(uintptr_t)lds_dst;
  asm volatile("global_load_async_to_lds_b128 %0, %1, off"
               :: "v"(lds_off), "v"(gsrc) : "memory");
}
__device__ __forceinline__ void wait_asynccnt0() {
  asm volatile("s_wait_asynccnt 0x0" ::: "memory");
}

// ---------------------------------------------------------------------------
// Prep: e_norm[k] = ||emb[k]||^2 ; pack emb into bf16 hi/lo pairs ; zero accums
// ---------------------------------------------------------------------------
__global__ __launch_bounds__(128) void vq_prep_kernel(
    const float* __restrict__ emb,
    unsigned* __restrict__ emb_hi, unsigned* __restrict__ emb_lo,
    float* __restrict__ e_norm, int* __restrict__ counts,
    double* __restrict__ loss_acc)
{
  __shared__ float red[128];
  const int m = blockIdx.x;
  const int t = threadIdx.x;
  float x0 = emb[m * E_DIM + 2 * t];
  float x1 = emb[m * E_DIM + 2 * t + 1];
  unsigned short h0 = f2bf(x0), h1 = f2bf(x1);
  unsigned short l0 = f2bf(x0 - bf2f(h0)), l1 = f2bf(x1 - bf2f(h1));
  emb_hi[m * 128 + t] = (unsigned)h0 | ((unsigned)h1 << 16);
  emb_lo[m * 128 + t] = (unsigned)l0 | ((unsigned)l1 << 16);
  red[t] = x0 * x0 + x1 * x1;
  __syncthreads();
  for (int off = 64; off > 0; off >>= 1) {
    if (t < off) red[t] += red[t + off];
    __syncthreads();
  }
  if (t == 0) {
    e_norm[m] = red[0];
    counts[m] = 0;
    if (m == 0) *loss_acc = 0.0;
  }
}

// ---------------------------------------------------------------------------
// Main kernel
// ---------------------------------------------------------------------------
__global__ __launch_bounds__(256) void vq_main_kernel(
    const float* __restrict__ z, const float* __restrict__ emb,
    const unsigned* __restrict__ emb_hi, const unsigned* __restrict__ emb_lo,
    const float* __restrict__ e_norm,
    float* __restrict__ zq_out, int* __restrict__ counts,
    double* __restrict__ loss_acc)
{
  // z tile, bf16 hi/lo, packed channel pairs: [128 pairs][64 cols]
  __shared__ unsigned zpk_hi[128 * NCOL];
  __shared__ unsigned zpk_lo[128 * NCOL];
  // double-buffered emb tile (32 codes x 128 pairs each)
  __shared__ unsigned apk_hi[2][32 * 128];
  __shared__ unsigned apk_lo[2][32 * 128];
  __shared__ float    enorm_all[N_E];
  __shared__ float    redv[2][64];
  __shared__ int      redi[2][64];
  __shared__ int      finalIdx[64];
  __shared__ float    lred[256];

  const int tid      = threadIdx.x;
  const int lane     = tid & 31;
  const int wave     = tid >> 5;
  const int code_row = wave >> 2;        // 0..1 : which 16-code row of the 32
  const int coltile  = wave & 3;         // 0..3 : which 16-col tile of the 64
  const int laneHi   = lane >> 4;        // 0|1  : half-wave
  const int lane16   = lane & 15;
  const int b        = blockIdx.x >> 4;
  const int n0       = (blockIdx.x & 15) * NCOL;

  const float* zb = z + (size_t)b * E_DIM * HW + n0;   // zb[c*HW + col]

  // kick off async copy of the first emb tile (overlaps z staging below)
#pragma unroll
  for (int j = 0; j < 4; ++j) {
    int off = tid * 4 + 1024 * j;
    async_ld_b128(&apk_hi[0][off], &emb_hi[off]);
    async_ld_b128(&apk_lo[0][off], &emb_lo[off]);
  }

  // preload all ||e||^2 once
  for (int i = tid; i < N_E; i += 256) enorm_all[i] = e_norm[i];

  // stage z tile once: split fp32 -> bf16 hi + bf16 residual, pack pairs
  for (int i = tid; i < 128 * NCOL; i += 256) {
    int p = i >> 6, col = i & 63;
    float x0 = zb[(size_t)(2 * p) * HW + col];
    float x1 = zb[(size_t)(2 * p + 1) * HW + col];
    unsigned short h0 = f2bf(x0), h1 = f2bf(x1);
    unsigned short l0 = f2bf(x0 - bf2f(h0)), l1 = f2bf(x1 - bf2f(h1));
    zpk_hi[i] = (unsigned)h0 | ((unsigned)h1 << 16);
    zpk_lo[i] = (unsigned)l0 | ((unsigned)l1 << 16);
  }

  float best = 3.402823466e+38f;
  int bestidx = 0;

  union U16 { unsigned u[8]; v16bf v; };

  const int m_lds  = code_row * 16 + lane16;   // emb row this lane supplies (A)
  const int colAll = coltile * 16 + lane16;    // z column this lane owns (B/D)

  for (int it = 0; it < NITER; ++it) {
    const int mbase = it * 32;
    const int cur = it & 1;
    wait_asynccnt0();       // this wave's async copy of tile `it` is in LDS
    __syncthreads();        // ...and every other wave has waited too

    // prefetch next emb tile into the other buffer (overlaps WMMA below)
    if (it + 1 < NITER) {
      const unsigned gbase = (unsigned)(it + 1) * 32 * 128;
#pragma unroll
      for (int j = 0; j < 4; ++j) {
        int off = tid * 4 + 1024 * j;
        async_ld_b128(&apk_hi[cur ^ 1][off], &emb_hi[gbase + off]);
        async_ld_b128(&apk_lo[cur ^ 1][off], &emb_lo[gbase + off]);
      }
    }

    v8f acc = {};
#pragma unroll
    for (int s = 0; s < 8; ++s) {              // K = 256 in steps of 32
      U16 ah, al, bh, bl;
      const int pbase = s * 16;
#pragma unroll
      for (int v = 0; v < 8; ++v) {
        // A (16-bit 16x32 layout): vgpr v<4 -> K pairs {2v,2v+1}(+8 hi-half),
        // vgpr v>=4 -> +16.
        int pa = pbase + ((v < 4) ? (v + laneHi * 4) : (8 + (v - 4) + laneHi * 4));
        ah.u[v] = apk_hi[cur][m_lds * 128 + pa];
        al.u[v] = apk_lo[cur][m_lds * 128 + pa];
        // B (32x16): same K order as A (vgpr v<4 -> pairs {2v,2v+1}(+8 hi-half), v>=4 -> +16)
        int pb = pbase + ((v < 4) ? (v + laneHi * 4) : (8 + (v - 4) + laneHi * 4));
        bh.u[v] = zpk_hi[pb * 64 + colAll];
        bl.u[v] = zpk_lo[pb * 64 + colAll];
      }
      // split-precision product: hi*hi + hi*lo + lo*hi, one f32 accumulator
      acc = __builtin_amdgcn_wmma_f32_16x16x32_bf16(false, ah.v, false, bh.v,
                                                    (short)0, acc, false, false);
      acc = __builtin_amdgcn_wmma_f32_16x16x32_bf16(false, ah.v, false, bl.v,
                                                    (short)0, acc, false, false);
      acc = __builtin_amdgcn_wmma_f32_16x16x32_bf16(false, al.v, false, bh.v,
                                                    (short)0, acc, false, false);
    }

    // D layout: vgpr v -> tile row v (lanes 0-15) / v+8 (lanes 16-31), col=lane16
#pragma unroll
    for (int v = 0; v < 8; ++v) {
      int mloc = code_row * 16 + laneHi * 8 + v;
      float sc = enorm_all[mbase + mloc] - 2.0f * acc[v];
      int code = mbase + mloc;
      if (sc < best) { best = sc; bestidx = code; }  // ascending codes => first-min
    }
  }

  // combine the two half-waves (they hold different codes of the same column)
  {
    float ob = __shfl_xor(best, 16, 32);
    int   oi = __shfl_xor(bestidx, 16, 32);
    if (ob < best || (ob == best && oi < bestidx)) { best = ob; bestidx = oi; }
  }
  __syncthreads();
  if (lane < 16) {
    redv[code_row][coltile * 16 + lane16] = best;
    redi[code_row][coltile * 16 + lane16] = bestidx;
  }
  __syncthreads();
  if (tid < 64) {
    float v0 = redv[0][tid], v1 = redv[1][tid];
    int   i0 = redi[0][tid], i1 = redi[1][tid];
    int fi = (v1 < v0 || (v1 == v0 && i1 < i0)) ? i1 : i0;
    finalIdx[tid] = fi;
    atomicAdd(&counts[fi], 1);
  }
  __syncthreads();

  // fused gather (straight-through output == z_q) + loss partial
  float lsum = 0.0f;
  float* outb = zq_out + (size_t)b * E_DIM * HW + n0;
  for (int i = tid; i < E_DIM * NCOL; i += 256) {
    int c = i >> 6, col = i & 63;
    int id = finalIdx[col];
    float e  = emb[(size_t)id * E_DIM + c];
    float zz = zb[(size_t)c * HW + col];
    float d  = e - zz;
    lsum += d * d;
    outb[(size_t)c * HW + col] = e;
  }
  lred[tid] = lsum;
  __syncthreads();
  for (int off = 128; off > 0; off >>= 1) {
    if (tid < off) lred[tid] += lred[tid + off];
    __syncthreads();
  }
  if (tid == 0) atomicAdd(loss_acc, (double)lred[0]);
}

// ---------------------------------------------------------------------------
// Finalize: loss scalar + perplexity from counts
// ---------------------------------------------------------------------------
__global__ __launch_bounds__(1024) void vq_finalize_kernel(
    const int* __restrict__ counts, const double* __restrict__ loss_acc,
    float* __restrict__ out)
{
  __shared__ float red[1024];
  const int t = threadIdx.x;
  float em = (float)counts[t] / (float)NPIX;
  red[t] = em * logf(em + 1e-10f);
  __syncthreads();
  for (int off = 512; off > 0; off >>= 1) {
    if (t < off) red[t] += red[t + off];
    __syncthreads();
  }
  if (t == 0) {
    // loss = (1 + beta) * mean((z_q - z)^2), beta = 0.25
    out[8388608] = 1.25f * (float)(*loss_acc / 8388608.0);
    out[8388609] = expf(-red[0]);
  }
}

// ---------------------------------------------------------------------------
extern "C" void kernel_launch(void* const* d_in, const int* in_sizes, int n_in,
                              void* d_out, int out_size, void* d_ws, size_t ws_size,
                              hipStream_t stream)
{
  const float* z   = (const float*)d_in[0];
  const float* emb = (const float*)d_in[1];
  float* out = (float*)d_out;

  // workspace layout (~1.03 MB)
  char* ws = (char*)d_ws;
  double*   loss_acc = (double*)ws;                         // 8 B
  float*    e_norm   = (float*)(ws + 8);                    // 4 KB
  int*      counts   = (int*)(ws + 8 + 4096);               // 4 KB
  unsigned* emb_hi   = (unsigned*)(ws + 8 + 8192);          // 512 KB
  unsigned* emb_lo   = emb_hi + (size_t)N_E * 128;          // 512 KB

  vq_prep_kernel<<<N_E, 128, 0, stream>>>(emb, emb_hi, emb_lo, e_norm, counts,
                                          loss_acc);
  vq_main_kernel<<<BATCH * (HW / NCOL), 256, 0, stream>>>(
      z, emb, emb_hi, emb_lo, e_norm, out, counts, loss_acc);
  vq_finalize_kernel<<<1, 1024, 0, stream>>>(counts, loss_acc, out);
}
